// NonLocalAttention_9972914061961
// MI455X (gfx1250) — matrix operand order repaired and checked by reading, not compiled, not run on hardware
//
#include <hip/hip_runtime.h>
#include <hip/hip_bf16.h>

typedef __attribute__((ext_vector_type(16))) __bf16 v16bf;
typedef __attribute__((ext_vector_type(8)))  __bf16 v8bf;
typedef __attribute__((ext_vector_type(8)))  float  v8f;
typedef int v4i_ __attribute__((vector_size(16)));   // matches builtin prototype

#define CH_  256
#define CR_  128
#define HW_  4096
#define NB_  4

#if defined(__has_builtin)
#if __has_builtin(__builtin_amdgcn_global_load_async_to_lds_b128) && \
    __has_builtin(__builtin_amdgcn_s_wait_asynccnt)
#define NLA_HAVE_ASYNC_LDS 1
#endif
#endif

__device__ __forceinline__ v16bf pack16(v8bf lo, v8bf hi) {
    union { v16bf v; v8bf h[2]; } u;
    u.h[0] = lo; u.h[1] = hi;
    return u.v;
}

// Copy 16 bytes global -> LDS. Async (ASYNCcnt-tracked, no VGPR bounce) when
// the toolchain exposes the gfx1250 builtin; plain load/store otherwise.
__device__ __forceinline__ void copy16_g2l(const __bf16* __restrict__ g,
                                           __bf16* __restrict__ l) {
#if defined(NLA_HAVE_ASYNC_LDS)
    __builtin_amdgcn_global_load_async_to_lds_b128(
        (__attribute__((address_space(1))) v4i_*)(g),
        (__attribute__((address_space(3))) v4i_*)(l), 0, 0);
#else
    *(v8bf*)l = *(const v8bf*)g;
#endif
}

__device__ __forceinline__ void wait_async_then_barrier() {
#if defined(NLA_HAVE_ASYNC_LDS)
    __builtin_amdgcn_s_wait_asynccnt(0);
#endif
    __syncthreads();
}

// ---------------------------------------------------------------------------
// Projection kernel: out[p,o] = PReLU( sum_c X[p,c] * W[o,c] + b[o] )
// grid.x = N*HW/128 (pixel blocks), grid.y = 32 (8 Q tiles, 8 K tiles, 16 V tiles)
// Q,K stored row-major [n*HW + p][CR] bf16; V stored col-major [n*CH + c][HW] bf16.
// ---------------------------------------------------------------------------
__global__ __launch_bounds__(256)
void nla_proj_kernel(const float* __restrict__ x,
                     const float* __restrict__ w1, const float* __restrict__ b1, const float* __restrict__ a1,
                     const float* __restrict__ w2, const float* __restrict__ b2, const float* __restrict__ a2,
                     const float* __restrict__ wa, const float* __restrict__ ba, const float* __restrict__ aa,
                     __bf16* __restrict__ Qb, __bf16* __restrict__ Kb, __bf16* __restrict__ Vt)
{
    __shared__ __bf16 xs[32 * 128];     // 32 channels x 128 pixels, bf16 (8 KB)

    const int tid  = threadIdx.x;
    const int wave = tid >> 5;
    const int lane = tid & 31;
    const int lo   = lane & 15;
    const int hi   = lane >> 4;

    const int pbase = blockIdx.x * 128;          // 128 divides HW -> single batch
    const int n     = pbase / HW_;
    const int pn    = pbase % HW_;
    const int oy    = blockIdx.y;

    const float* W; const float* Bv; const float* Av; int obase, mode;
    if (oy < 8)       { W = w1; Bv = b1; Av = a1; obase = oy * 16;        mode = 0; }
    else if (oy < 16) { W = w2; Bv = b2; Av = a2; obase = (oy - 8) * 16;  mode = 1; }
    else              { W = wa; Bv = ba; Av = aa; obase = (oy - 16) * 16; mode = 2; }

    const float* xn = x + (size_t)n * CH_ * HW_;
    const int srow = tid >> 3;          // 0..31  (channel within chunk)
    const int scol = (tid & 7) * 16;    // 0..112 (pixel within tile)

    v8f acc = {};
    for (int kb = 0; kb < CH_; kb += 32) {
        // cooperative stage: 32 channels x 128 pixels -> LDS bf16
        const float* src = xn + (size_t)(kb + srow) * HW_ + pn + scol;
        #pragma unroll
        for (int i = 0; i < 4; ++i) {
            float4 v = ((const float4*)src)[i];
            __bf16* d = &xs[srow * 128 + scol + i * 4];
            d[0] = (__bf16)v.x; d[1] = (__bf16)v.y; d[2] = (__bf16)v.z; d[3] = (__bf16)v.w;
        }
        __syncthreads();

        // A fragment (16 pixels x 32 channels), k = 16*hi + j (contiguous order)
        v16bf af;
        #pragma unroll
        for (int j = 0; j < 16; ++j) {
            int c = (hi << 4) + j;
            af[j] = xs[c * 128 + wave * 16 + lo];
        }
        // B fragment (32 channels x 16 outputs): W[o, c] contiguous over c
        v16bf bfr;
        const float* wp = W + (size_t)(obase + lo) * CH_ + kb + hi * 16;
        #pragma unroll
        for (int i = 0; i < 4; ++i) {
            float4 v = ((const float4*)wp)[i];
            bfr[i*4+0] = (__bf16)v.x; bfr[i*4+1] = (__bf16)v.y;
            bfr[i*4+2] = (__bf16)v.z; bfr[i*4+3] = (__bf16)v.w;
        }
        acc = __builtin_amdgcn_wmma_f32_16x16x32_bf16(false, af, false, bfr,
                                                      (short)0, acc, false, false);
        __syncthreads();
    }

    const float bias  = Bv[obase + lo];
    const float alpha = Av[0];
    #pragma unroll
    for (int r = 0; r < 8; ++r) {
        float v = acc[r] + bias;
        v = (v >= 0.0f) ? v : alpha * v;            // PReLU
        const int p = pn + wave * 16 + r + (hi << 3);
        const int o = obase + lo;
        if (mode == 0)      Qb[((size_t)n * HW_ + p) * CR_ + o] = (__bf16)v;
        else if (mode == 1) Kb[((size_t)n * HW_ + p) * CR_ + o] = (__bf16)v;
        else                Vt[((size_t)n * CH_ + o) * HW_ + p] = (__bf16)v;
    }
}

// ---------------------------------------------------------------------------
// Flash attention: S = Q K^T (no scale), P = softmax_row(S), O = P V, out = O + x
// grid.x = N*HW/128; 8 waves/block; each wave owns 16 query rows.
// K/V tiles staged into LDS once per block via async global->LDS copies and
// shared by all 8 waves; key tile = 64: 16 QK + 32 PV WMMAs per iter per wave.
// ---------------------------------------------------------------------------
__global__ __launch_bounds__(256)
void nla_attn_kernel(const float* __restrict__ x,
                     const __bf16* __restrict__ Qb, const __bf16* __restrict__ Kb,
                     const __bf16* __restrict__ Vt, float* __restrict__ out)
{
    __shared__ __bf16 ks[64 * 128];        // K tile:  [key][channel]  (16 KB)
    __shared__ __bf16 vsl[256 * 64];       // V tile:  [channel][key]  (32 KB)
    __shared__ __bf16 pls[8 * 16 * 64];    // per-wave P bounce        (16 KB)

    const int tid  = threadIdx.x;
    const int wave = tid >> 5;
    const int lane = tid & 31;
    const int lo   = lane & 15;
    const int hi   = lane >> 4;

    const int n    = blockIdx.x >> 5;
    const int qb   = (blockIdx.x & 31) * 128;
    const int row0 = qb + wave * 16;
    const size_t nrow = (size_t)n * HW_;

    // Q fragments: 16 rows x 128 channels, 4 K-steps of 32 (held in VGPRs)
    v16bf qf[4];
    #pragma unroll
    for (int kb = 0; kb < 4; ++kb) {
        const __bf16* p = Qb + (nrow + row0 + lo) * (size_t)CR_ + kb * 32 + hi * 16;
        qf[kb] = pack16(*(const v8bf*)p, *(const v8bf*)(p + 8));
    }

    v8f ov[16] = {};                       // O accumulators: 16 channel tiles
    float mrow[8], lrow[8];
    #pragma unroll
    for (int r = 0; r < 8; ++r) { mrow[r] = -1e30f; lrow[r] = 0.0f; }

    __bf16* myp = &pls[wave * (16 * 64)];

    for (int kt = 0; kt < HW_; kt += 64) {
        // ---- stage K (64x128) and V (256x64) tiles into LDS, async ----
        #pragma unroll
        for (int i = 0; i < 4; ++i) {                  // 1024 x 16B chunks
            const int chunk = tid + i * 256;
            const int row = chunk >> 4;                // key row 0..63
            const int c8  = (chunk & 15) * 8;          // channel offset
            copy16_g2l(Kb + (nrow + kt + row) * (size_t)CR_ + c8,
                       &ks[row * 128 + c8]);
        }
        #pragma unroll
        for (int i = 0; i < 8; ++i) {                  // 2048 x 16B chunks
            const int chunk = tid + i * 256;
            const int c  = chunk >> 3;                 // channel 0..255
            const int k8 = (chunk & 7) * 8;            // key offset
            copy16_g2l(Vt + ((size_t)n * CH_ + c) * HW_ + kt + k8,
                       &vsl[c * 64 + k8]);
        }
        // prefetch next key tile of K toward L2 while we compute
        if (kt + 64 < HW_)
            __builtin_prefetch(Kb + (nrow + kt + 64 + lo) * (size_t)CR_, 0, 1);
        wait_async_then_barrier();

        // ---- S = Q K^T for this 16x64 tile (B-fragments from LDS) ----
        v8f s[4] = {};
        #pragma unroll
        for (int ct = 0; ct < 4; ++ct) {
            #pragma unroll
            for (int kb = 0; kb < 4; ++kb) {
                v16bf bk = *(const v16bf*)&ks[(ct * 16 + lo) * 128 + kb * 32 + hi * 16];
                s[ct] = __builtin_amdgcn_wmma_f32_16x16x32_bf16(false, qf[kb], false, bk,
                                                                (short)0, s[ct], false, false);
            }
        }

        // ---- online softmax over key dimension ----
        #pragma unroll
        for (int r = 0; r < 8; ++r) {
            float tm = fmaxf(fmaxf(s[0][r], s[1][r]), fmaxf(s[2][r], s[3][r]));
            tm = fmaxf(tm, __shfl_xor(tm, 1, 32));
            tm = fmaxf(tm, __shfl_xor(tm, 2, 32));
            tm = fmaxf(tm, __shfl_xor(tm, 4, 32));
            tm = fmaxf(tm, __shfl_xor(tm, 8, 32));
            const float mn = fmaxf(mrow[r], tm);
            const float sc = __expf(mrow[r] - mn);
            mrow[r] = mn;
            float ps = 0.0f;
            #pragma unroll
            for (int ct = 0; ct < 4; ++ct) {
                float e = __expf(s[ct][r] - mn);
                s[ct][r] = e;
                ps += e;
            }
            ps += __shfl_xor(ps, 1, 32);
            ps += __shfl_xor(ps, 2, 32);
            ps += __shfl_xor(ps, 4, 32);
            ps += __shfl_xor(ps, 8, 32);
            lrow[r] = lrow[r] * sc + ps;
            #pragma unroll
            for (int nt = 0; nt < 16; ++nt) ov[nt][r] *= sc;
        }

        // ---- C/D-layout P -> A-layout via per-wave LDS bounce ----
        #pragma unroll
        for (int ct = 0; ct < 4; ++ct)
            #pragma unroll
            for (int r = 0; r < 8; ++r)
                myp[(r + hi * 8) * 64 + ct * 16 + lo] = (__bf16)s[ct][r];

        v16bf pf[2];
        #pragma unroll
        for (int kk = 0; kk < 2; ++kk) {
            const __bf16* p = myp + lo * 64 + kk * 32 + hi * 16;
            pf[kk] = pack16(*(const v8bf*)p, *(const v8bf*)(p + 8));
        }

        // ---- O += P V (B-fragments from LDS) ----
        #pragma unroll
        for (int nt = 0; nt < 16; ++nt) {
            #pragma unroll
            for (int kk = 0; kk < 2; ++kk) {
                v16bf bv = *(const v16bf*)&vsl[(nt * 16 + lo) * 64 + kk * 32 + hi * 16];
                ov[nt] = __builtin_amdgcn_wmma_f32_16x16x32_bf16(false, pf[kk], false, bv,
                                                                 (short)0, ov[nt], false, false);
            }
        }
        __syncthreads();     // protect ks/vsl before next iteration's staging
    }

    // ---- normalize, residual, store ----
    #pragma unroll
    for (int nt = 0; nt < 16; ++nt) {
        const int c = nt * 16 + lo;
        const float* xr = x   + ((size_t)n * CH_ + c) * HW_;
        float* orow     = out + ((size_t)n * CH_ + c) * HW_;
        #pragma unroll
        for (int r = 0; r < 8; ++r) {
            const int p = row0 + r + hi * 8;
            orow[p] = ov[nt][r] / lrow[r] + xr[p];   // RESCALE = 1.0
        }
    }
}

extern "C" void kernel_launch(void* const* d_in, const int* in_sizes, int n_in,
                              void* d_out, int out_size, void* d_ws, size_t ws_size,
                              hipStream_t stream)
{
    const float* x  = (const float*)d_in[0];
    const float* w1 = (const float*)d_in[1];
    const float* b1 = (const float*)d_in[2];
    const float* a1 = (const float*)d_in[3];
    const float* w2 = (const float*)d_in[4];
    const float* b2 = (const float*)d_in[5];
    const float* a2 = (const float*)d_in[6];
    const float* wa = (const float*)d_in[7];
    const float* ba = (const float*)d_in[8];
    const float* aa = (const float*)d_in[9];
    float* out = (float*)d_out;

    // workspace layout: Q (4 MB) | K (4 MB) | Vt (8 MB), all bf16, 32B aligned
    __bf16* Qb = (__bf16*)d_ws;
    __bf16* Kb = Qb + (size_t)NB_ * HW_ * CR_;
    __bf16* Vt = Kb + (size_t)NB_ * HW_ * CR_;

    dim3 gp(NB_ * HW_ / 128, 32);
    nla_proj_kernel<<<gp, 256, 0, stream>>>(x, w1, b1, a1, w2, b2, a2, wa, ba, aa,
                                            Qb, Kb, Vt);
    nla_attn_kernel<<<NB_ * HW_ / 128, 256, 0, stream>>>(x, Qb, Kb, Vt, out);
}
